// FFTGADBase_15126874817207
// MI455X (gfx1250) — matrix operand order repaired and checked by reading, not compiled, not run on hardware
//
#include <hip/hip_runtime.h>
#include <math.h>

#define BATCH 2
#define H 1024
#define W 1024
#define SH 128
#define SW 128
#define L_DIFF 0.24f
#define K_EDGE 0.03f
#define EPSV 1e-8f

typedef __attribute__((ext_vector_type(16))) _Float16 v16h;
typedef __attribute__((ext_vector_type(8)))  float    v8f;

#if defined(__has_builtin)
#if __has_builtin(__builtin_amdgcn_tensor_load_to_lds)
#define HAVE_TDM 1
#endif
#if __has_builtin(__builtin_amdgcn_global_load_async_to_lds_b32)
#define HAVE_ASYNC 1
#endif
#endif
#ifndef HAVE_TDM
#define HAVE_TDM 0
#endif
#ifndef HAVE_ASYNC
#define HAVE_ASYNC 0
#endif

__device__ __forceinline__ float gfun(float x) {
  return 1.0f / (1.0f + x * x * (1.0f / (K_EDGE * K_EDGE)));
}

#if HAVE_TDM
typedef unsigned int u32x4 __attribute__((ext_vector_type(4)));
typedef int i32x8 __attribute__((ext_vector_type(8)));
typedef int i32x4 __attribute__((ext_vector_type(4)));

__device__ __forceinline__ unsigned lds_offset_of(const void* p) {
  return (unsigned)(unsigned long long)(const __attribute__((address_space(3))) void*)p;
}

// Build D# (groups 0/1 per CDNA5 ISA) and issue a
// 2D f32 tile load: tile_w x tile_h from gsrc (row stride stride_elems) into
// contiguous LDS at lds_off. 2D: groups 2/3 zero.
__device__ __forceinline__ void tdm_load_2d_f32(const float* gsrc,
                                                unsigned lds_off, int tile_w,
                                                int tile_h, int stride_elems,
                                                int tensor_w, int tensor_h) {
  unsigned long long ga = (unsigned long long)gsrc;
  unsigned td0 = (unsigned)tensor_w, td1 = (unsigned)tensor_h;
  unsigned long long s0 = (unsigned long long)(unsigned)stride_elems;
  unsigned long long s1 = s0 * (unsigned long long)td1;
  u32x4 g0;
  g0[0] = 1u;                                              // count=1 (user D#)
  g0[1] = lds_off;                                         // lds_addr
  g0[2] = (unsigned)(ga & 0xffffffffull);                  // global_addr lo
  g0[3] = (unsigned)((ga >> 32) & 0x1ffffffull) | (2u << 30);  // hi | type=2
  i32x8 g1;
  g1[0] = (int)(2u << 16);                                 // data_size=4B
  g1[1] = (int)(td0 << 16);                                // tensor_dim0[15:0]
  g1[2] = (int)((td0 >> 16) | (td1 << 16));                // dim0 hi | dim1 lo
  g1[3] = (int)((td1 >> 16) | ((unsigned)tile_w << 16));   // dim1 hi | tile_dim0
  g1[4] = (int)((unsigned)tile_h);                         // tile_dim1, tile_dim2=0
  g1[5] = (int)(unsigned)(s0 & 0xffffffffull);             // dim0_stride[31:0]
  g1[6] = (int)(((unsigned)((s0 >> 32) & 0xffffull)) |
                ((unsigned)(s1 & 0xffffull) << 16));       // d0s hi | d1s lo
  g1[7] = (int)((s1 >> 16) & 0xffffffffull);               // d1s[47:16]
  i32x4 z4 = {0, 0, 0, 0};
  i32x8 z8 = {0, 0, 0, 0, 0, 0, 0, 0};
  __builtin_amdgcn_tensor_load_to_lds(g0, g1, z4, z4, z8, 0);
}
#endif

__device__ __forceinline__ void wait_async_done() {
#if HAVE_ASYNC
#if __has_builtin(__builtin_amdgcn_s_wait_asynccnt)
  __builtin_amdgcn_s_wait_asynccnt(0);
#else
  asm volatile("s_wait_asynccnt 0" ::: "memory");
#endif
#endif
}

// ---------------------------------------------------------------------------
// K1: edge-stopping coefficients cv/ch from feats = concat(guide, y_bicubic),
// also seeds the working image buffer and the cv/ch output regions.
// ---------------------------------------------------------------------------
__global__ __launch_bounds__(256) void coeff_init_kernel(
    const float* __restrict__ guide, const float* __restrict__ ybic,
    float* __restrict__ cv, float* __restrict__ ch, float* __restrict__ imgA,
    float* __restrict__ ocv, float* __restrict__ och) {
  size_t idx = (size_t)blockIdx.x * blockDim.x + threadIdx.x;
  if (idx >= (size_t)BATCH * H * W) return;
  int x = (int)(idx % W);
  int y = (int)((idx / W) % H);
  int b = (int)(idx / ((size_t)H * W));
  const float* gb = guide + (size_t)b * 3 * H * W;
  const float* ib = ybic + (size_t)b * H * W;
  float f0 = gb[(size_t)y * W + x];
  float f1 = gb[(size_t)H * W + (size_t)y * W + x];
  float f2 = gb[2 * (size_t)H * W + (size_t)y * W + x];
  float f3 = ib[(size_t)y * W + x];
  imgA[idx] = f3;
  if (y < H - 1) {
    float m = 0.25f * (fabsf(gb[(size_t)(y + 1) * W + x] - f0) +
                       fabsf(gb[(size_t)H * W + (size_t)(y + 1) * W + x] - f1) +
                       fabsf(gb[2 * (size_t)H * W + (size_t)(y + 1) * W + x] - f2) +
                       fabsf(ib[(size_t)(y + 1) * W + x] - f3));
    float g = gfun(m);
    size_t o = (size_t)b * (H - 1) * W + (size_t)y * W + x;
    cv[o] = g;
    ocv[o] = g;
  }
  if (x < W - 1) {
    float m = 0.25f * (fabsf(gb[(size_t)y * W + x + 1] - f0) +
                       fabsf(gb[(size_t)H * W + (size_t)y * W + x + 1] - f1) +
                       fabsf(gb[2 * (size_t)H * W + (size_t)y * W + x + 1] - f2) +
                       fabsf(ib[(size_t)y * W + x + 1] - f3));
    float g = gfun(m);
    size_t o = (size_t)b * H * (W - 1) + (size_t)y * (W - 1) + x;
    ch[o] = g;
    och[o] = g;
  }
}

// ---------------------------------------------------------------------------
// WMMA helpers: CDNA5 wave32 16x16x32 f16 -> f32, fragment layouts per
// CDNA5 ISA 7.12.2.
// ---------------------------------------------------------------------------
__device__ __forceinline__ v8f wmma16(v16h a, v16h b, v8f c) {
  return __builtin_amdgcn_wmma_f32_16x16x32_f16(false, a, false, b, (short)0, c,
                                                false, false);
}

__device__ __forceinline__ v16h load_fragA(const _Float16* M, int m0, int k0,
                                           int lane, float sgn) {
  v16h r;
  int m = m0 + (lane & 15);
  int half = (lane >> 4) & 1;
#pragma unroll
  for (int v = 0; v < 8; ++v) {
    int basek = (v < 4) ? (2 * v) : (16 + 2 * (v - 4));
#pragma unroll
    for (int s = 0; s < 2; ++s) {
      int k = k0 + basek + 8 * half + s;
      r[2 * v + s] = (_Float16)((float)M[m * 64 + k] * sgn);
    }
  }
  return r;
}

__device__ __forceinline__ v16h load_fragB(const _Float16* M, int k0, int n0,
                                           int lane) {
  v16h r;
  int n = n0 + (lane & 15);
  int half = (lane >> 4) & 1;
#pragma unroll
  for (int v = 0; v < 8; ++v) {
#pragma unroll
    for (int s = 0; s < 2; ++s) {
      int k = k0 + ((v < 4) ? (2 * v) : (16 + 2 * (v - 4))) + 8 * half + s;
      r[2 * v + s] = M[k * 64 + n];
    }
  }
  return r;
}

__device__ void mm64(const _Float16* A1, const _Float16* B1, float s1,
                     const _Float16* A2, const _Float16* B2, float s2,
                     _Float16* C, int tid) {
  int lane = tid & 31;
  int wid = tid >> 5;
  for (int t = wid; t < 16; t += 8) {
    int m0 = (t >> 2) << 4, n0 = (t & 3) << 4;
    v8f acc = {};
    for (int kt = 0; kt < 2; ++kt) {
      int k0 = kt << 5;
      acc = wmma16(load_fragA(A1, m0, k0, lane, s1),
                   load_fragB(B1, k0, n0, lane), acc);
      if (A2)
        acc = wmma16(load_fragA(A2, m0, k0, lane, s2),
                     load_fragB(B2, k0, n0, lane), acc);
    }
    int col = n0 + (lane & 15);
    int rbase = m0 + (((lane >> 4) & 1) << 3);
#pragma unroll
    for (int v = 0; v < 8; ++v) C[(rbase + v) * 64 + col] = (_Float16)acc[v];
  }
}

// ---------------------------------------------------------------------------
// K2: blockwise spectral diffusion via 64x64 DFT as WMMA matmuls.
// Image tile staged through LDS with async global->LDS loads (ASYNCcnt),
// overlapped with cos/sin table generation.
// ---------------------------------------------------------------------------
__global__ __launch_bounds__(256) void fft_block_kernel(
    float* __restrict__ img, const float* __restrict__ cv,
    const float* __restrict__ ch, int colY, int colX) {
  __shared__ _Float16 Bt[4096], Ctab[4096], Stab[4096];
  __shared__ _Float16 T12[8192];  // T1|T2, also aliased as f32 staging tile
  __shared__ _Float16 T3[4096], T4[4096];
  __shared__ float sred[6];
  _Float16* T1 = T12;
  _Float16* T2 = T12 + 4096;
  int tid = threadIdx.x;
  int byi = 2 * blockIdx.y + colY;
  int bxi = 2 * blockIdx.x + colX;
  int b = blockIdx.z;
  int y0 = byi * 48, x0 = bxi * 48;
  if (y0 + 64 > H || x0 + 64 > W) return;  // partial edge blocks: leave as-is

  if (tid < 6) sred[tid] = 0.0f;
  __syncthreads();

  const float* cvb = cv + (size_t)b * (H - 1) * W;
  const float* chb = ch + (size_t)b * H * (W - 1);

  // block stats: cv/ch means and uniform-region fraction (3x3 variance).
  float cvsum = 0.f, chsum = 0.f, cvc = 0.f, chc = 0.f, uc = 0.f, ud = 0.f;
  for (int e = tid; e < 4096; e += 256) {
    int ry = e >> 6, rx = e & 63;
    int gy = y0 + ry, gx = x0 + rx;
    if (gy < H - 1) { cvsum += cvb[(size_t)gy * W + gx]; cvc += 1.0f; }
    if (gx < W - 1) { chsum += chb[(size_t)gy * (W - 1) + gx]; chc += 1.0f; }
    if (gy < H - 1 && gx < W - 1) {
      float s1 = 0.f, s2 = 0.f, t1 = 0.f, t2 = 0.f;
#pragma unroll
      for (int dy = -1; dy <= 1; ++dy)
#pragma unroll
        for (int dx = -1; dx <= 1; ++dx) {
          int yy = gy + dy, xx = gx + dx;
          if (yy >= 0 && yy < H - 1 && xx >= 0 && xx < W) {
            float v = cvb[(size_t)yy * W + xx]; s1 += v; s2 += v * v;
          }
          if (yy >= 0 && yy < H && xx >= 0 && xx < W - 1) {
            float v = chb[(size_t)yy * (W - 1) + xx]; t1 += v; t2 += v * v;
          }
        }
      float mv = s1 * (1.0f / 9.0f), vv = s2 * (1.0f / 9.0f) - mv * mv;
      float mh = t1 * (1.0f / 9.0f), vh = t2 * (1.0f / 9.0f) - mh * mh;
      uc += (vv < 0.1f && vh < 0.1f) ? 1.0f : 0.0f;
      ud += 1.0f;
    }
  }
  atomicAdd(&sred[0], cvsum); atomicAdd(&sred[1], chsum);
  atomicAdd(&sred[2], cvc);   atomicAdd(&sred[3], chc);
  atomicAdd(&sred[4], uc);    atomicAdd(&sred[5], ud);
  __syncthreads();
  float cvm = sred[0] / fmaxf(sred[2], 1.0f);
  float chm = sred[1] / fmaxf(sred[3], 1.0f);
  float unif = sred[4] / fmaxf(sred[5], 1.0f);
  if (!(unif > 0.7f)) return;

  float* imb = img + (size_t)b * H * W;
  float* stage = (float*)T12;  // 4096 f32 staging tile over T1|T2

#if HAVE_ASYNC
  // async global->LDS: no VGPR round-trip; overlap with table generation.
  for (int e = tid; e < 4096; e += 256) {
    int ry = e >> 6, rx = e & 63;
    const float* gp = &imb[(size_t)(y0 + ry) * W + (x0 + rx)];
    __builtin_amdgcn_global_load_async_to_lds_b32(
        (__attribute__((address_space(1))) int*)gp,
        (__attribute__((address_space(3))) int*)&stage[e], 0, 0);
  }
  for (int e = tid; e < 4096; e += 256) {
    int ry = e >> 6, rx = e & 63;
    float ang = 6.28318530718f * (float)((ry * rx) & 63) * (1.0f / 64.0f);
    Ctab[e] = (_Float16)__cosf(ang);
    Stab[e] = (_Float16)__sinf(ang);
  }
  wait_async_done();
  __syncthreads();
  for (int e = tid; e < 4096; e += 256) Bt[e] = (_Float16)stage[e];
  __syncthreads();
#else
  for (int e = tid; e < 4096; e += 256) {
    int ry = e >> 6, rx = e & 63;
    float ang = 6.28318530718f * (float)((ry * rx) & 63) * (1.0f / 64.0f);
    Ctab[e] = (_Float16)__cosf(ang);
    Stab[e] = (_Float16)__sinf(ang);
    Bt[e] = (_Float16)imb[(size_t)(y0 + ry) * W + (x0 + rx)];
  }
  __syncthreads();
#endif

  mm64(Ctab, Bt, 1.0f, nullptr, nullptr, 0.0f, T1, tid);   // P
  mm64(Stab, Bt, 1.0f, nullptr, nullptr, 0.0f, T2, tid);   // Q
  __syncthreads();
  mm64(T1, Ctab, 1.0f, T2, Stab, -1.0f, T3, tid);          // Xr
  mm64(T1, Stab, -1.0f, T2, Ctab, -1.0f, T4, tid);         // Xi
  __syncthreads();

  for (int e = tid; e < 4096; e += 256) {  // spectral scale lambda^10
    int fy = e >> 6, fx = e & 63;
    float lam =
        1.0f -
        2.0f * L_DIFF * cvm * (1.0f - __cosf(6.28318530718f * fy * (1.0f / 64.0f))) -
        2.0f * L_DIFF * chm * (1.0f - __cosf(6.28318530718f * fx * (1.0f / 64.0f)));
    float l2 = lam * lam, l4 = l2 * l2, l8 = l4 * l4, l10 = l8 * l2;
    T3[e] = (_Float16)((float)T3[e] * l10);
    T4[e] = (_Float16)((float)T4[e] * l10);
  }
  __syncthreads();

  mm64(T3, Ctab, 1.0f, T4, Stab, -1.0f, T1, tid);          // Zr
  mm64(T3, Stab, 1.0f, T4, Ctab, 1.0f, T2, tid);           // Zi
  __syncthreads();

  // out = (C*Zr - S*Zi)/4096; blend directly from WMMA accumulators.
  int lane = tid & 31, wid = tid >> 5;
  for (int t = wid; t < 16; t += 8) {
    int m0 = (t >> 2) << 4, n0 = (t & 3) << 4;
    v8f acc = {};
    for (int kt = 0; kt < 2; ++kt) {
      int k0 = kt << 5;
      acc = wmma16(load_fragA(Ctab, m0, k0, lane, 1.0f),
                   load_fragB(T1, k0, n0, lane), acc);
      acc = wmma16(load_fragA(Stab, m0, k0, lane, -1.0f),
                   load_fragB(T2, k0, n0, lane), acc);
    }
    int col = n0 + (lane & 15);
    int rbase = m0 + (((lane >> 4) & 1) << 3);
#pragma unroll
    for (int v = 0; v < 8; ++v) {
      int row = rbase + v;
      size_t off = (size_t)(y0 + row) * W + (x0 + col);
      float oldv = imb[off];
      float nv = acc[v] * (1.0f / 4096.0f);
      float byf = (y0 > 0 && row < 16) ? (float)row * (1.0f / 16.0f) : 1.0f;
      float bxf = (x0 > 0 && col < 16) ? (float)col * (1.0f / 16.0f) : 1.0f;
      float bl = byf * bxf;
      imb[off] = oldv + (nv - oldv) * bl;
    }
  }
}

// ---------------------------------------------------------------------------
// K3: fused diffuse+adjust iteration. Interior tiles stage their three LDS
// tiles (image+halo, cv, ch) with the Tensor Data Mover (one D# each, issued
// by wave 0, waited with s_wait_tensorcnt); edge tiles fall back to scalar
// zero-filled loads. 8x8 cell means via ds_add_f32.
// ---------------------------------------------------------------------------
__global__ __launch_bounds__(256) void diffuse_adjust_kernel(
    const float* __restrict__ Iin, float* __restrict__ Iout,
    const float* __restrict__ cv, const float* __restrict__ ch,
    const float* __restrict__ source, const float* __restrict__ mask_lr) {
  __shared__ float sI[66 * 66];
  __shared__ float sCV[65 * 64];
  __shared__ float sCH[64 * 65];
  __shared__ float cellsum[64];
  __shared__ float ratio[64];
  int tid = threadIdx.x;
  int b = blockIdx.z;
  int y0 = blockIdx.y << 6, x0 = blockIdx.x << 6;
  const float* Ib = Iin + (size_t)b * H * W;
  const float* cvb = cv + (size_t)b * (H - 1) * W;
  const float* chb = ch + (size_t)b * H * (W - 1);

  if (tid == 0) {
    __builtin_prefetch(source + (size_t)b * SH * SW + (size_t)(y0 >> 3) * SW + (x0 >> 3), 0, 1);
    __builtin_prefetch(mask_lr + (size_t)b * SH * SW + (size_t)(y0 >> 3) * SW + (x0 >> 3), 0, 1);
  }

  bool interior = (y0 > 0) && (y0 + 65 <= H) && (x0 > 0) && (x0 + 65 <= W);
#if HAVE_TDM
  if (interior) {
    if (tid < 32) {  // wave 0 issues three TDM tile loads
      tdm_load_2d_f32(&Ib[(size_t)(y0 - 1) * W + (x0 - 1)], lds_offset_of(sI),
                      66, 66, W, W, H);
      tdm_load_2d_f32(&cvb[(size_t)(y0 - 1) * W + x0], lds_offset_of(sCV),
                      64, 65, W, W, H - 1);
      tdm_load_2d_f32(&chb[(size_t)y0 * (W - 1) + (x0 - 1)], lds_offset_of(sCH),
                      65, 64, W - 1, W - 1, H);
      __builtin_amdgcn_s_wait_tensorcnt(0);
    }
  } else
#else
  (void)interior;
#endif
  {
    for (int e = tid; e < 66 * 66; e += 256) {
      int r = e / 66, c = e % 66;
      int gy = y0 - 1 + r, gx = x0 - 1 + c;
      sI[e] = (gy >= 0 && gy < H && gx >= 0 && gx < W) ? Ib[(size_t)gy * W + gx] : 0.0f;
    }
    for (int e = tid; e < 65 * 64; e += 256) {
      int r = e >> 6, c = e & 63;
      int gy = y0 - 1 + r, gx = x0 + c;
      sCV[e] = (gy >= 0 && gy < H - 1) ? cvb[(size_t)gy * W + gx] : 0.0f;
    }
    for (int e = tid; e < 64 * 65; e += 256) {
      int r = e / 65, c = e % 65;
      int gy = y0 + r, gx = x0 - 1 + c;
      sCH[e] = (gx >= 0 && gx < W - 1) ? chb[(size_t)gy * (W - 1) + gx] : 0.0f;
    }
  }
  if (tid < 64) cellsum[tid] = 0.0f;
  __syncthreads();

  float vnew[16];
  int j = 0;
  for (int e = tid; e < 4096; e += 256, ++j) {
    int ry = e >> 6, rx = e & 63;
    int r = ry + 1, c = rx + 1;
    float Ic = sI[r * 66 + c];
    float fl = L_DIFF * (sCV[(ry + 1) * 64 + rx] * (sI[(r + 1) * 66 + c] - Ic) -
                         sCV[ry * 64 + rx] * (Ic - sI[(r - 1) * 66 + c]) +
                         sCH[ry * 65 + rx + 1] * (sI[r * 66 + c + 1] - Ic) -
                         sCH[ry * 65 + rx] * (Ic - sI[r * 66 + c - 1]));
    float In = Ic + fl;
    vnew[j] = In;
    atomicAdd(&cellsum[(ry >> 3) * 8 + (rx >> 3)], In);
  }
  __syncthreads();
  if (tid < 64) {
    int gcy = (y0 >> 3) + (tid >> 3), gcx = (x0 >> 3) + (tid & 7);
    size_t so = (size_t)b * SH * SW + (size_t)gcy * SW + gcx;
    float mean = cellsum[tid] * (1.0f / 64.0f);
    ratio[tid] = (mask_lr[so] < 0.5f) ? 1.0f : source[so] / (mean + EPSV);
  }
  __syncthreads();
  float* Ob = Iout + (size_t)b * H * W;
  j = 0;
  for (int e = tid; e < 4096; e += 256, ++j) {
    int ry = e >> 6, rx = e & 63;
    Ob[(size_t)(y0 + ry) * W + (x0 + rx)] = vnew[j] * ratio[(ry >> 3) * 8 + (rx >> 3)];
  }
}

// ---------------------------------------------------------------------------
// Host-side orchestration (all launches on `stream`; graph-capture safe).
// ---------------------------------------------------------------------------
extern "C" void kernel_launch(void* const* d_in, const int* in_sizes, int n_in,
                              void* d_out, int out_size, void* d_ws,
                              size_t ws_size, hipStream_t stream) {
  (void)in_sizes; (void)n_in; (void)out_size; (void)ws_size;
  const float* guide = (const float*)d_in[0];     // (2,3,1024,1024)
  const float* source = (const float*)d_in[1];    // (2,1,128,128)
  const float* mask_lr = (const float*)d_in[2];   // (2,1,128,128)
  const float* ybic = (const float*)d_in[3];      // (2,1,1024,1024)
  float* out = (float*)d_out;                     // y_pred | cv | ch

  const size_t NIMG = (size_t)BATCH * H * W;
  const size_t NCV = (size_t)BATCH * (H - 1) * W;

  float* imgA = (float*)d_ws;
  float* imgB = imgA + NIMG;
  float* cvw = imgB + NIMG;
  float* chw = cvw + NCV;
  float* out_cv = out + NIMG;
  float* out_ch = out_cv + NCV;

  coeff_init_kernel<<<dim3((unsigned)((NIMG + 255) / 256)), 256, 0, stream>>>(
      guide, ybic, cvw, chw, imgA, out_cv, out_ch);

  for (int cy = 0; cy < 2; ++cy)
    for (int cx = 0; cx < 2; ++cx)
      fft_block_kernel<<<dim3(11, 11, BATCH), 256, 0, stream>>>(imgA, cvw, chw,
                                                                cy, cx);

  dim3 dgrid(W / 64, H / 64, BATCH);
  for (int it = 0; it < 500; ++it) {
    const float* src = (it & 1) ? imgB : imgA;
    float* dst = (it == 499) ? out : ((it & 1) ? imgA : imgB);
    diffuse_adjust_kernel<<<dgrid, 256, 0, stream>>>(src, dst, cvw, chw, source,
                                                     mask_lr);
  }
}
